// SMA_76596446757465
// MI455X (gfx1250) — matrix operand order repaired and checked by reading, not compiled, not run on hardware
//
#include <hip/hip_runtime.h>
#include <stdint.h>

// ---------------- problem constants ----------------
#define T_DIM 16
#define B_DIM 2
#define C_DIM 128
#define H_DIM 64
#define W_DIM 64
#define S_DIM 4
#define HW    4096                  // H*W
#define CHW   (C_DIM*HW)            // 524288
#define TB    (T_DIM*B_DIM)         // 32
#define BCHW  (B_DIM*CHW)           // 1048576
#define TOT   (T_DIM*BCHW)          // 16777216
#define CR_   8
#define TR_   4
#define EPSB  1e-5f

// ---------------- workspace layout (bytes) ----------------
#define OFF_XB     0ull             // bf16 NHWC activations:  33,554,432 B
#define OFF_WPK    33554432ull      // packed bf16 weights:     6,422,528 B
#define WPK_STRIDE 802816ull        // halfs per branch (49 taps worst case)
#define OFF_Y      39976960ull      // conv+BN out (per-s reuse): 67,108,864 B
#define OFF_SPK    107085824ull     // u8 spikes [S][T][B][C][H][W]: 67,108,864 B
#define OFF_FM     174194688ull     // f32 [T][B][C]
#define OFF_CW     (OFF_FM + 16384ull)     // f32 [S][T][B][C]
#define OFF_TW     (OFF_CW + 65536ull)     // f32 [S][T][B]
#define OFF_TMASK  (OFF_TW + 512ull)       // i32 [S][T]
#define OFF_CMASK  (OFF_TMASK + 256ull)    // i32 [S][C]

typedef __attribute__((ext_vector_type(16))) __bf16 v16bf;
typedef __attribute__((ext_vector_type(8)))  float  v8f;

union ABfrag { v16bf v; uint4 q[2]; };

__device__ __forceinline__ uint16_t f32_to_bf16(float f) {
    uint32_t u = __float_as_uint(f);
    u += 0x7FFFu + ((u >> 16) & 1u);   // round-to-nearest-even
    return (uint16_t)(u >> 16);
}

// x f32 [TB][C][H][W] -> xb bf16 [TB][H][W][C]
__global__ void k_convert_nhwc(const float* __restrict__ x, uint16_t* __restrict__ xb) {
    int idx = blockIdx.x * 256 + threadIdx.x;    // [0, TOT)
    int n  = idx / CHW;
    int r  = idx - n * CHW;
    int c  = r >> 12;
    int hw = r & 4095;
    xb[(size_t)(n * HW + hw) * C_DIM + c] = f32_to_bf16(x[idx]);
}

// pack conv weights [Cout][Cin][k][k] f32 into WMMA B-fragment order:
// wpk[((tap*4+kb)*8+nb)*32 + lane][j] = w[nb*16 + lane%16][kb*32 + (lane/16)*16 + j][kh][kw]
__global__ void k_pack_w(const float* __restrict__ w, uint16_t* __restrict__ wpk, int k) {
    int idx = blockIdx.x * 256 + threadIdx.x;
    int total = k * k * 16384;
    if (idx >= total) return;
    int j    = idx & 15;
    int lane = (idx >> 4) & 31;
    int nb   = (idx >> 9) & 7;
    int kb   = (idx >> 12) & 3;
    int tap  = idx >> 14;
    int cout = nb * 16 + (lane & 15);
    int cin  = kb * 32 + (lane >> 4) * 16 + j;
    int kh = tap / k, kw = tap - kh * k;
    float v = w[((size_t)(cout * C_DIM + cin) * k + kh) * k + kw];
    wpk[idx] = f32_to_bf16(v);
}

// implicit-GEMM conv (SAME) + fused bias/BN, templated on kernel size K.
// One block = 16-pixel row tile of one image row; wave nb owns cout block
// [nb*16, nb*16+16). A slab staged in LDS once per block (zero-filled halo),
// A fragments via ds_load_b128, B via packed global b128, bf16 WMMA with
// fp32 accumulate on two alternating accumulators for ILP.
template <int K>
__global__ void __launch_bounds__(256) k_conv_wmma(
        const uint16_t* __restrict__ xb, const uint16_t* __restrict__ wpk,
        const float* __restrict__ gamma, const float* __restrict__ beta,
        const float* __restrict__ mean,  const float* __restrict__ var,
        const float* __restrict__ bias,  float* __restrict__ y) {
    constexpr int PAD   = K / 2;
    constexpr int WSLAB = 16 + K - 1;             // pixels per LDS row
    constexpr int NCHUNK = K * WSLAB * 16;        // 16B chunks in slab
    __shared__ uint16_t smA[K * WSLAB * C_DIM];

    int tile = blockIdx.x;                 // 32*64*4 = 8192 blocks
    int w0 = (tile & 3) * 16;
    int h  = (tile >> 2) & 63;
    int n  = tile >> 8;
    int lane = threadIdx.x & 31;
    int nb   = threadIdx.x >> 5;
    int hi   = lane >> 4;                  // K-half selector

    // ---- cooperative stage: [K rows][WSLAB px][128 ch] with zero halo ----
    for (int q = threadIdx.x; q < NCHUNK; q += 256) {
        int row = q / (WSLAB * 16);
        int rem = q - row * (WSLAB * 16);
        int px  = rem >> 4;
        int seg = rem & 15;
        int hp = h + row - PAD;
        int wq = w0 + px - PAD;
        uint4 val = make_uint4(0u, 0u, 0u, 0u);
        if (hp >= 0 && hp < H_DIM && wq >= 0 && wq < W_DIM)
            val = *reinterpret_cast<const uint4*>(
                xb + ((size_t)(n * H_DIM + hp) * W_DIM + wq) * C_DIM + seg * 8);
        *reinterpret_cast<uint4*>(&smA[q * 8]) = val;
    }
    __syncthreads();

    // ---- fully unrolled K*K*4 WMMA chain ----
    v8f acc0 = {}, acc1 = {};
    const int apx = (lane & 15);           // A-row pixel within tile
    #pragma unroll
    for (int kh = 0; kh < K; ++kh) {
        #pragma unroll
        for (int kw = 0; kw < K; ++kw) {
            #pragma unroll
            for (int kb = 0; kb < 4; ++kb) {
                ABfrag a, b;
                // A k order chosen here (contiguous): low lanes K 0-15, high lanes K 16-31 (matches packed B)
                const uint16_t* ap =
                    &smA[((kh * WSLAB) + apx + kw) * C_DIM + kb * 32 + hi * 16];
                a.q[0] = *reinterpret_cast<const uint4*>(ap);
                a.q[1] = *reinterpret_cast<const uint4*>(ap + 8);
                size_t b0 = ((size_t)(((kh * K + kw) * 4 + kb) * 8 + nb) * 32 + lane) * 16;
                b.q[0] = *reinterpret_cast<const uint4*>(wpk + b0);
                b.q[1] = *reinterpret_cast<const uint4*>(wpk + b0 + 8);
                if (kb & 1)
                    acc1 = __builtin_amdgcn_wmma_f32_16x16x32_bf16(
                        false, a.v, false, b.v, (short)0, acc1, false, false);
                else
                    acc0 = __builtin_amdgcn_wmma_f32_16x16x32_bf16(
                        false, a.v, false, b.v, (short)0, acc0, false, false);
            }
        }
    }
    v8f acc = acc0 + acc1;

    // ---- fused bias + BN epilogue: y = conv*scale + shift (streaming NT) ----
    int cout = nb * 16 + (lane & 15);
    float sc = gamma[cout] * rsqrtf(var[cout] + EPSB);
    float sh = (bias[cout] - mean[cout]) * sc + beta[cout];
    size_t ob = ((size_t)(n * C_DIM + cout) * H_DIM + h) * W_DIM + w0 + hi * 8;
    #pragma unroll
    for (int r = 0; r < 8; ++r)
        __builtin_nontemporal_store(acc[r] * sc + sh, &y[ob + r]);
}

// LIF: v = (v+x)/2 ; spike = v>=1 ; hard reset to 0.
__global__ void k_lif(const float* __restrict__ y, uint8_t* __restrict__ spk) {
    size_t j = (size_t)blockIdx.x * 256 + threadIdx.x;   // [0, BCHW)
    float v = 0.f;
    #pragma unroll
    for (int t = 0; t < T_DIM; ++t) {
        float xv = __builtin_nontemporal_load(&y[(size_t)t * BCHW + j]);
        v = 0.5f * (v + xv);
        bool sp = (v >= 1.0f);
        spk[(size_t)t * BCHW + j] = sp ? (uint8_t)1 : (uint8_t)0;
        if (sp) v = 0.f;
    }
}

// fm[t,b,c] = (1/HW) * sum_s sum_hw spikes
__global__ void k_fm(const uint8_t* __restrict__ spk, float* __restrict__ fm) {
    int blk = blockIdx.x;                  // (t*B+b)*C + c
    size_t o = (size_t)blk * HW;
    int sum = 0;
    for (int s = 0; s < S_DIM; ++s) {
        const uint8_t* p = spk + (size_t)s * TOT + o;
        for (int i = threadIdx.x; i < HW; i += 256) sum += p[i];
    }
    __shared__ int red[256];
    red[threadIdx.x] = sum;
    __syncthreads();
    for (int st = 128; st > 0; st >>= 1) {
        if (threadIdx.x < st) red[threadIdx.x] += red[threadIdx.x + st];
        __syncthreads();
    }
    if (threadIdx.x == 0) fm[blk] = (float)red[0] * (1.0f / HW);
}

// all attention math + AZO mask computation (tiny; single block)
__global__ void __launch_bounds__(256) k_attn(
        const float* __restrict__ fm,
        const float* __restrict__ cb_w, const float* __restrict__ cb_b,
        const float* __restrict__ ca_w, const float* __restrict__ ca_b,
        const float* __restrict__ tb_w, const float* __restrict__ tb_b,
        const float* __restrict__ ta_w, const float* __restrict__ ta_b,
        float* __restrict__ cw, float* __restrict__ tw,
        int* __restrict__ tmask, int* __restrict__ cmask) {
    __shared__ float sm_mid[256];    // [t][b][r]
    __shared__ float sm_tpool[32];   // [b][t]
    __shared__ float sm_tmid[8];     // [b][r]
    __shared__ float sm_tw[128];     // [s][t][b]
    __shared__ float sm_twsum[64];   // [s][t]
    __shared__ float sm_kall[512];   // [s][c]
    int tid = threadIdx.x;

    { // mid = relu(fm @ cb_w^T + cb_b)
        int t = tid >> 4, b = (tid >> 3) & 1, r = tid & 7;
        const float* f = fm + (t * B_DIM + b) * C_DIM;
        float acc = cb_b[r];
        for (int c = 0; c < C_DIM; ++c) acc += f[c] * cb_w[r * C_DIM + c];
        sm_mid[tid] = fmaxf(acc, 0.f);
    }
    __syncthreads();
    // cw (pre-softmax) [s][t][b][c]
    for (int u = 0; u < 64; ++u) {
        int e = tid * 64 + u;
        int s = e >> 12, t = (e >> 8) & 15, b = (e >> 7) & 1, c = e & 127;
        const float* m  = sm_mid + (t * B_DIM + b) * CR_;
        const float* aw = ca_w + (size_t)(s * C_DIM + c) * CR_;
        float acc = ca_b[s * C_DIM + c];
        #pragma unroll
        for (int r = 0; r < CR_; ++r) acc += m[r] * aw[r];
        cw[e] = acc;
    }
    __syncthreads();
    // softmax over s
    for (int idx = tid; idx < 4096; idx += 256) {
        float v0 = cw[idx], v1 = cw[4096 + idx], v2 = cw[8192 + idx], v3 = cw[12288 + idx];
        float mx = fmaxf(fmaxf(v0, v1), fmaxf(v2, v3));
        float e0 = expf(v0 - mx), e1 = expf(v1 - mx), e2 = expf(v2 - mx), e3 = expf(v3 - mx);
        float inv = 1.f / (e0 + e1 + e2 + e3);
        cw[idx] = e0 * inv; cw[4096 + idx] = e1 * inv;
        cw[8192 + idx] = e2 * inv; cw[12288 + idx] = e3 * inv;
    }
    __syncthreads();
    if (tid < 32) { // tpool[b][t] = mean_c fm
        int b = tid >> 4, t = tid & 15;
        const float* f = fm + (t * B_DIM + b) * C_DIM;
        float acc = 0.f;
        for (int c = 0; c < C_DIM; ++c) acc += f[c];
        sm_tpool[b * 16 + t] = acc * (1.0f / C_DIM);
    }
    __syncthreads();
    if (tid < 8) { // tmid[b][r]
        int b = tid >> 2, r = tid & 3;
        float acc = tb_b[r];
        for (int t = 0; t < T_DIM; ++t) acc += sm_tpool[b * 16 + t] * tb_w[r * T_DIM + t];
        sm_tmid[b * 4 + r] = fmaxf(acc, 0.f);
    }
    __syncthreads();
    if (tid < 128) { // tw pre-softmax [s][t][b]
        int s = tid >> 5, t = (tid >> 1) & 15, b = tid & 1;
        float acc = ta_b[s * T_DIM + t];
        #pragma unroll
        for (int r = 0; r < TR_; ++r) acc += sm_tmid[b * 4 + r] * ta_w[(s * T_DIM + t) * TR_ + r];
        sm_tw[tid] = acc;
    }
    __syncthreads();
    if (tid < 32) { // softmax over s + export
        float v0 = sm_tw[tid], v1 = sm_tw[32 + tid], v2 = sm_tw[64 + tid], v3 = sm_tw[96 + tid];
        float mx = fmaxf(fmaxf(v0, v1), fmaxf(v2, v3));
        float e0 = expf(v0 - mx), e1 = expf(v1 - mx), e2 = expf(v2 - mx), e3 = expf(v3 - mx);
        float inv = 1.f / (e0 + e1 + e2 + e3);
        sm_tw[tid] = e0 * inv; sm_tw[32 + tid] = e1 * inv;
        sm_tw[64 + tid] = e2 * inv; sm_tw[96 + tid] = e3 * inv;
        tw[tid] = sm_tw[tid]; tw[32 + tid] = sm_tw[32 + tid];
        tw[64 + tid] = sm_tw[64 + tid]; tw[96 + tid] = sm_tw[96 + tid];
    }
    __syncthreads();
    if (tid < 64) { // twsum[s][t] over b
        int s = tid >> 4, t = tid & 15;
        sm_twsum[tid] = sm_tw[s * 32 + t * 2] + sm_tw[s * 32 + t * 2 + 1];
    }
    __syncthreads();
    if (tid < S_DIM) { // T_min: 4 smallest (ties -> lowest index, like top_k)
        int s = tid;
        bool used[16];
        for (int t = 0; t < 16; ++t) used[t] = false;
        for (int it = 0; it < 4; ++it) {
            int best = 0; float bv = 3.4e38f;
            for (int t = 0; t < 16; ++t)
                if (!used[t] && sm_twsum[s * 16 + t] < bv) { bv = sm_twsum[s * 16 + t]; best = t; }
            used[best] = true;
        }
        for (int t = 0; t < 16; ++t) tmask[s * 16 + t] = used[t] ? 1 : 0;
    }
    __syncthreads();
    for (int e = tid; e < 512; e += 256) { // k_all[s][c]
        int s = e >> 7, c = e & 127;
        float acc = 0.f;
        for (int t = 0; t < 16; ++t)
            if (tmask[s * 16 + t])
                acc += cw[((s * 16 + t) * 2 + 0) * 128 + c]
                     + cw[((s * 16 + t) * 2 + 1) * 128 + c];
        sm_kall[e] = acc;
    }
    __syncthreads();
    if (tid < S_DIM) { // C_min: 32 smallest channels
        int s = tid;
        bool used[128];
        for (int c = 0; c < 128; ++c) used[c] = false;
        for (int it = 0; it < 32; ++it) {
            int best = 0; float bv = 3.4e38f;
            for (int c = 0; c < 128; ++c)
                if (!used[c] && sm_kall[s * 128 + c] < bv) { bv = sm_kall[s * 128 + c]; best = c; }
            used[best] = true;
        }
        for (int c = 0; c < 128; ++c) cmask[s * 128 + c] = used[c] ? 1 : 0;
    }
}

// out[t,b,c,hw] = sum_s tw[s,ts,b]*cw[s,ts,b,c]*spk[s,ts,b,c,hw]
// ts = t-1 iff (t in T_min[s]) && (c in C_min[s]) && t>0 (AZO replacement)
__global__ void k_final(const uint8_t* __restrict__ spk,
                        const float* __restrict__ cw, const float* __restrict__ tw,
                        const int* __restrict__ tmask, const int* __restrict__ cmask,
                        float* __restrict__ out) {
    int blk = blockIdx.x;              // (t*B+b)*C + c
    int t = blk >> 8, b = (blk >> 7) & 1, c = blk & 127;
    float cf[S_DIM];
    const uint8_t* sp[S_DIM];
    #pragma unroll
    for (int s = 0; s < S_DIM; ++s) {
        int repl = tmask[s * 16 + t] & cmask[s * 128 + c];
        int ts = (repl && t > 0) ? (t - 1) : t;
        cf[s] = tw[(s * 16 + ts) * 2 + b] * cw[((s * 16 + ts) * 2 + b) * 128 + c];
        sp[s] = spk + (size_t)s * TOT + ((size_t)(ts * 2 + b) * 128 + c) * HW;
    }
    size_t ob = (size_t)blk * HW;
    for (int i = threadIdx.x; i < HW; i += 256) {
        float acc = 0.f;
        #pragma unroll
        for (int s = 0; s < S_DIM; ++s) acc += cf[s] * (float)sp[s][i];
        out[ob + i] = acc;
    }
}

extern "C" void kernel_launch(void* const* d_in, const int* in_sizes, int n_in,
                              void* d_out, int out_size, void* d_ws, size_t ws_size,
                              hipStream_t stream) {
    (void)in_sizes; (void)n_in; (void)out_size; (void)ws_size;
    const float* x = (const float*)d_in[0];
    const float* wconv[4] = {(const float*)d_in[1], (const float*)d_in[2],
                             (const float*)d_in[3], (const float*)d_in[4]};
    const float* bconv[4] = {(const float*)d_in[5], (const float*)d_in[6],
                             (const float*)d_in[7], (const float*)d_in[8]};
    const float* gamma = (const float*)d_in[9];
    const float* beta  = (const float*)d_in[10];
    const float* mean  = (const float*)d_in[11];
    const float* var   = (const float*)d_in[12];
    const float* cb_w  = (const float*)d_in[13];
    const float* cb_b  = (const float*)d_in[14];
    const float* ca_w  = (const float*)d_in[15];
    const float* ca_b  = (const float*)d_in[16];
    const float* tb_w  = (const float*)d_in[17];
    const float* tb_b  = (const float*)d_in[18];
    const float* ta_w  = (const float*)d_in[19];
    const float* ta_b  = (const float*)d_in[20];

    char* ws = (char*)d_ws;
    uint16_t* xb  = (uint16_t*)(ws + OFF_XB);
    uint16_t* wpk = (uint16_t*)(ws + OFF_WPK);
    float*    y   = (float*)(ws + OFF_Y);
    uint8_t*  spk = (uint8_t*)(ws + OFF_SPK);
    float*    fm  = (float*)(ws + OFF_FM);
    float*    cw  = (float*)(ws + OFF_CW);
    float*    tw  = (float*)(ws + OFF_TW);
    int*   tmaskp = (int*)(ws + OFF_TMASK);
    int*   cmaskp = (int*)(ws + OFF_CMASK);
    float* out    = (float*)d_out;

    static const int KS[4] = {1, 3, 5, 7};

    k_convert_nhwc<<<TOT / 256, 256, 0, stream>>>(x, xb);
    for (int s = 0; s < S_DIM; ++s) {
        int k = KS[s];
        int tot = k * k * 16384;
        k_pack_w<<<(tot + 255) / 256, 256, 0, stream>>>(wconv[s], wpk + (size_t)s * WPK_STRIDE, k);
    }
    for (int s = 0; s < S_DIM; ++s) {
        const uint16_t* wp = wpk + (size_t)s * WPK_STRIDE;
        const float *g = gamma + s * C_DIM, *be = beta + s * C_DIM,
                    *mu = mean + s * C_DIM, *va = var + s * C_DIM, *bi = bconv[s];
        switch (KS[s]) {
            case 1: k_conv_wmma<1><<<8192, 256, 0, stream>>>(xb, wp, g, be, mu, va, bi, y); break;
            case 3: k_conv_wmma<3><<<8192, 256, 0, stream>>>(xb, wp, g, be, mu, va, bi, y); break;
            case 5: k_conv_wmma<5><<<8192, 256, 0, stream>>>(xb, wp, g, be, mu, va, bi, y); break;
            default: k_conv_wmma<7><<<8192, 256, 0, stream>>>(xb, wp, g, be, mu, va, bi, y); break;
        }
        k_lif<<<BCHW / 256, 256, 0, stream>>>(y, spk + (size_t)s * TOT);
    }
    k_fm<<<4096, 256, 0, stream>>>(spk, fm);
    k_attn<<<1, 256, 0, stream>>>(fm, cb_w, cb_b, ca_w, ca_b, tb_w, tb_b, ta_w, ta_b,
                                  cw, tw, tmaskp, cmaskp);
    k_final<<<4096, 256, 0, stream>>>(spk, cw, tw, tmaskp, cmaskp, out);
}
